// qGPS_29532195127695
// MI455X (gfx1250) — compile-verified, matrix operand order checked by reading, not run on hardware
//
#include <hip/hip_runtime.h>

typedef _Float16 v16h __attribute__((ext_vector_type(16)));
typedef _Float16 v8h  __attribute__((ext_vector_type(8)));
typedef float    v8f  __attribute__((ext_vector_type(8)));

#define B_N 8192
#define L_N 256
#define M_N 64

// ---------------------------------------------------------------------------
// Prep: Delta[m,l] = log(eps[1,m,l]) - log(eps[0,m,l])  (f16)
//       C[m]      = sum_l log(eps[0,m,l])               (f32)
// eps layout: (D, M, L) row-major -> eps[d*M*L + m*L + l]
// ---------------------------------------------------------------------------
__global__ __launch_bounds__(256) void qgps_prep(const float* __restrict__ eps,
                                                 _Float16* __restrict__ dlt,
                                                 float* __restrict__ logc) {
    const int tid = threadIdx.x;
    for (int i = tid; i < M_N * L_N; i += 256) {
        float e0 = eps[i];               // d = 0
        float e1 = eps[M_N * L_N + i];   // d = 1
        dlt[i] = (_Float16)(__logf(e1) - __logf(e0));
    }
    if (tid < M_N) {
        float s = 0.f;
        for (int l = 0; l < L_N; ++l) s += __logf(eps[tid * L_N + l]);
        logc[tid] = s;
    }
}

// ---------------------------------------------------------------------------
// Main: out[b] = sum_m exp( C[m] + sum_l s[b,l] * Delta[m,l] )
// Block: 32 samples x all 64 modes. 8 waves, each owns one 16x16 (b x m) tile.
// Delta staged global->LDS via async-to-LDS (ASYNCcnt) while VALU converts S.
// ---------------------------------------------------------------------------
__global__ __launch_bounds__(256) void qgps_main(const int* __restrict__ inp,
                                                 const _Float16* __restrict__ dlt,
                                                 const float* __restrict__ logc,
                                                 float* __restrict__ out) {
    __shared__ _Float16 sS[32 * L_N];    // 16 KB: S tile, f16 {0,1}
    __shared__ _Float16 sD[M_N * L_N];   // 32 KB: Delta (shared by all waves)
    __shared__ float    sC[M_N];
    __shared__ float    sPart[32];

    const int tid = threadIdx.x;
    const int b0  = blockIdx.x * 32;

    // --- Async-copy Delta (32 KB) global -> LDS: 2048 B128 transfers, 8/thread.
    // LDS generic address: low 32 bits are the LDS byte offset (VDST operand).
    {
        unsigned ldsBase = (unsigned)(unsigned long long)(uintptr_t)sD;
        const char* gD = (const char*)dlt;
#pragma unroll
        for (int i = 0; i < 8; ++i) {
            unsigned byteOff = (unsigned)(tid + i * 256) * 16u;
            unsigned ldsAddr = ldsBase + byteOff;
            asm volatile(
                "global_load_async_to_lds_b128 %0, %1, %2 offset:0"
                :: "v"(ldsAddr), "v"(byteOff), "s"(gD)
                : "memory");
        }
    }

    // --- Stage S meanwhile: 8192 contiguous int32 loads (coalesced), cvt to f16.
    const int* gin = inp + b0 * L_N;
    for (int i = tid; i < 32 * L_N; i += 256)
        sS[i] = (_Float16)(float)gin[i];

    if (tid < M_N) sC[tid] = logc[tid];
    if (tid < 32)  sPart[tid] = 0.f;

    asm volatile("s_wait_asynccnt 0" ::: "memory");
    __syncthreads();

    const int lane    = tid & 31;
    const int w       = tid >> 5;
    const int rowHalf = (w & 4) ? 16 : 0;   // sample-row half of the block
    const int mt      = (w & 3) << 4;       // mode tile base

    // 16-bit A-matrix fragment layout (ISA 7.12.2):
    // lane<16 : elems 0..7 = K 0..7,  elems 8..15 = K 16..23
    // lane>=16: elems 0..7 = K 8..15, elems 8..15 = K 24..31
    const int half16 = (lane < 16) ? 0 : 8;
    const int r = rowHalf + (lane & 15);    // sample row within block
    const int n = mt + (lane & 15);         // mode column

    const _Float16* sRowA = sS + r * L_N;
    const _Float16* sRowB = sD + n * L_N;

    v8f acc = {};
#pragma unroll
    for (int k0 = 0; k0 < L_N; k0 += 32) {
        // 16-byte aligned LDS vector loads -> ds_load_b128
        v8h alo = *(const v8h*)(sRowA + k0 + half16);
        v8h ahi = *(const v8h*)(sRowA + k0 + 16 + half16);
        v8h blo = *(const v8h*)(sRowB + k0 + half16);
        v8h bhi = *(const v8h*)(sRowB + k0 + 16 + half16);
        v16h a = __builtin_shufflevector(alo, ahi, 0,1,2,3,4,5,6,7,8,9,10,11,12,13,14,15);
        v16h b = __builtin_shufflevector(blo, bhi, 0,1,2,3,4,5,6,7,8,9,10,11,12,13,14,15);
        // D = A(16x32 f16) x B(32x16 f16) + C(f32)
        acc = __builtin_amdgcn_wmma_f32_16x16x32_f16(
            /*neg_a=*/false, a, /*neg_b=*/false, b,
            /*c_mod=*/(short)0, acc, /*reuse_a=*/false, /*reuse_b=*/false);
    }

    // Epilogue. C/D layout: VGPR j, lane<16 -> (M=j, N=lane); lane>=16 -> (M=j+8, N=lane-16).
    const float cAdd = sC[mt + (lane & 15)];
#pragma unroll
    for (int j = 0; j < 8; ++j) {
        float v = __expf(acc[j] + cAdd);
        // reduce over the 16 mode-columns (stays inside each 16-lane half)
        v += __shfl_xor(v, 1, 32);
        v += __shfl_xor(v, 2, 32);
        v += __shfl_xor(v, 4, 32);
        v += __shfl_xor(v, 8, 32);
        if ((lane & 15) == 0) {
            int row = rowHalf + j + ((lane >= 16) ? 8 : 0);
            atomicAdd(&sPart[row], v);   // ds_add_f32: combine the 4 mode tiles
        }
    }
    __syncthreads();
    if (tid < 32) out[b0 + tid] = sPart[tid];
}

// ---------------------------------------------------------------------------
extern "C" void kernel_launch(void* const* d_in, const int* in_sizes, int n_in,
                              void* d_out, int out_size, void* d_ws, size_t ws_size,
                              hipStream_t stream) {
    (void)in_sizes; (void)n_in; (void)out_size; (void)ws_size;
    const int*   inp = (const int*)d_in[0];     // (B, L) int32 in {0,1}
    const float* eps = (const float*)d_in[1];   // (D, M, L) f32

    _Float16* dlt  = (_Float16*)d_ws;                                      // 32 KB
    float*    logc = (float*)((char*)d_ws + M_N * L_N * sizeof(_Float16)); // 256 B
    float*    outp = (float*)d_out;

    qgps_prep<<<1, 256, 0, stream>>>(eps, dlt, logc);
    qgps_main<<<B_N / 32, 256, 0, stream>>>(inp, dlt, logc, outp);
}
